// ConditionalMultiheadAttention_64768106824153
// MI455X (gfx1250) — matrix operand order repaired and checked by reading, not compiled, not run on hardware
//
#include <hip/hip_runtime.h>
#include <hip/hip_bf16.h>

// ---------------------------------------------------------------------------
// ConditionalMultiheadAttention for MI455X (gfx1250), wave32 + WMMA f16.
//   B=16, N=512, D=768, H=12, E=8, HD=64
// Wave tile: 32(M) x 64(N), K step 32 -> 8 v_wmma per K step.
// ---------------------------------------------------------------------------

#define B_ 16
#define N_ 512
#define D_ 768
#define H_ 12
#define E_ 8
#define HD_ 64

typedef __attribute__((ext_vector_type(16))) _Float16 v16h;
typedef __attribute__((ext_vector_type(8)))  _Float16 v8h;
typedef __attribute__((ext_vector_type(4)))  _Float16 v4h;
typedef __attribute__((ext_vector_type(8)))  float    v8f;
typedef __attribute__((ext_vector_type(4)))  float    v4f;

// ---- WMMA fragment loaders (CDNA5 ISA 7.12.2 layouts, wave32) --------------
// A fragment 16x32 f16: lane l holds row (l&15), K runs [(l>>4)*8 .. +7], +16.
__device__ __forceinline__ v16h load_a_f16(const _Float16* __restrict__ A,
                                           int lda, int lane) {
  const int m  = lane & 15;
  const int kb = (lane >> 4) * 8;
  const _Float16* p = A + (size_t)m * lda + kb;
  v8h lo = *(const v8h*)(p);
  v8h hi = *(const v8h*)(p + 16);
  v16h r;
#pragma unroll
  for (int j = 0; j < 8; ++j) { r[j] = lo[j]; r[j + 8] = hi[j]; }
  return r;
}

// A fragment from f32 source (attention probs), converted to f16 in-register.
__device__ __forceinline__ v16h load_a_f32cvt(const float* __restrict__ A,
                                              int lda, int lane) {
  const int m  = lane & 15;
  const int kb = (lane >> 4) * 8;
  const float* p = A + (size_t)m * lda + kb;
  v16h r;
#pragma unroll
  for (int j = 0; j < 8; ++j) {
    r[j]     = (_Float16)p[j];
    r[j + 8] = (_Float16)p[j + 16];
  }
  return r;
}

// B fragment 32x16 f16: lane l is column (l&15); K runs [(l>>4)*8 .. +7], +16.
// Bm points at a row-major matrix whose ROWS are output columns, inner dim = K.
__device__ __forceinline__ v16h load_b_f16(const _Float16* __restrict__ Bm,
                                           int ldb, int lane) {
  const int c  = lane & 15;
  const int kb = (lane >> 4) * 8;
  const _Float16* p = Bm + (size_t)c * ldb + kb;
  v8h lo = *(const v8h*)(p);
  v8h hi = *(const v8h*)(p + 16);
  v16h r;
#pragma unroll
  for (int j = 0; j < 8; ++j) { r[j] = lo[j]; r[j + 8] = hi[j]; }
  return r;
}

__device__ __forceinline__ v8f wmma16(v16h a, v16h b, v8f c) {
  return __builtin_amdgcn_wmma_f32_16x16x32_f16(false, a, false, b,
                                                (short)0, c, false, false);
}

// ---------------------------------------------------------------------------
// K0: combined expert weights  Wc[p][b][o][i] = sum_e g[b,e] * W_p[e][o][i]
// ---------------------------------------------------------------------------
__global__ void k_combine_w(const float* __restrict__ g,
                            const float* __restrict__ Wq,
                            const float* __restrict__ Wk,
                            const float* __restrict__ Wv,
                            _Float16* __restrict__ Wc) {
  const int p = blockIdx.y / B_;
  const int b = blockIdx.y % B_;
  const float* W = (p == 0) ? Wq : ((p == 1) ? Wk : Wv);
  _Float16* dst = Wc + ((size_t)p * B_ + b) * (size_t)D_ * D_;

  const int i4 = blockIdx.x * blockDim.x + threadIdx.x;   // 0 .. D*D/4-1
  if (i4 >= (D_ * D_) / 4) return;
  const size_t base = (size_t)i4 * 4;

  v4f acc = {0.f, 0.f, 0.f, 0.f};
#pragma unroll
  for (int e = 0; e < E_; ++e) {
    const float ge = g[b * E_ + e];
    v4f w = *(const v4f*)(W + (size_t)e * D_ * D_ + base);
    acc += ge * w;
  }
  v4h o;
#pragma unroll
  for (int j = 0; j < 4; ++j) o[j] = (_Float16)acc[j];
  *(v4h*)(dst + base) = o;
}

// combined biases bc[p][b][o]
__global__ void k_combine_b(const float* __restrict__ g,
                            const float* __restrict__ bq,
                            const float* __restrict__ bk,
                            const float* __restrict__ bv,
                            float* __restrict__ bc) {
  const int tid = blockIdx.x * blockDim.x + threadIdx.x;
  if (tid >= 3 * B_ * D_) return;
  const int p = tid / (B_ * D_);
  const int r = tid % (B_ * D_);
  const int b = r / D_;
  const int o = r % D_;
  const float* src = (p == 0) ? bq : ((p == 1) ? bk : bv);
  float acc = 0.f;
#pragma unroll
  for (int e = 0; e < E_; ++e) acc += g[b * E_ + e] * src[e * D_ + o];
  bc[tid] = acc;
}

// generic f32 -> f16 convert (float4 granularity)
__global__ void k_cvt_f16(const float* __restrict__ src,
                          _Float16* __restrict__ dst, int n4) {
  const int i = blockIdx.x * blockDim.x + threadIdx.x;
  if (i >= n4) return;
  v4f s = *(const v4f*)(src + (size_t)i * 4);
  v4h o;
#pragma unroll
  for (int j = 0; j < 4; ++j) o[j] = (_Float16)s[j];
  *(v4h*)(dst + (size_t)i * 4) = o;
}

// ---------------------------------------------------------------------------
// G1: QKV projection.  C[n,o] = sum_i x16[b,n,i] * Wc[p][b][o,i] + bc[p][b][o]
// wave -> 32(M) x 64(N) tile.  grid: x = 384 blocks (8 waves), y = p.
// V (p==2) stored transposed: Vt[b][d][n]
// ---------------------------------------------------------------------------
__global__ void k_qkv_gemm(const _Float16* __restrict__ x16,
                           const _Float16* __restrict__ Wc,
                           const float* __restrict__ bc,
                           _Float16* __restrict__ Q16,
                           _Float16* __restrict__ K16,
                           _Float16* __restrict__ Vt16) {
  const int p    = blockIdx.y;
  const int lane = threadIdx.x & 31;
  const int wid  = blockIdx.x * 8 + (threadIdx.x >> 5);   // 0 .. 3071
  const int b  = wid / 192;            // 192 = 16 Mtiles * 12 Ntiles
  const int rm = wid % 192;
  const int mt = rm / 12;              // 0..15   (rows of 32)
  const int nt = rm % 12;              // 0..11   (cols of 64)

  const _Float16* Abase = x16 + (size_t)b * N_ * D_ + (size_t)mt * 32 * D_;
  const _Float16* Wbase = Wc + ((size_t)p * B_ + b) * (size_t)D_ * D_
                             + (size_t)nt * 64 * D_;
  const float* bcp = bc + ((size_t)p * B_ + b) * D_ + nt * 64;

  v8f acc[2][4] = {};
  for (int k0 = 0; k0 < D_; k0 += 32) {
    v16h a0 = load_a_f16(Abase + k0, D_, lane);
    v16h a1 = load_a_f16(Abase + (size_t)16 * D_ + k0, D_, lane);
#pragma unroll
    for (int t = 0; t < 4; ++t) {
      v16h bf = load_b_f16(Wbase + (size_t)t * 16 * D_ + k0, D_, lane);
      acc[0][t] = wmma16(a0, bf, acc[0][t]);
      acc[1][t] = wmma16(a1, bf, acc[1][t]);
    }
  }

  const int cl = lane & 15;
#pragma unroll
  for (int half = 0; half < 2; ++half) {
    const int rbase = mt * 32 + half * 16 + (lane >> 4) * 8;
#pragma unroll
    for (int t = 0; t < 4; ++t) {
      const int c = nt * 64 + t * 16 + cl;
      const float bias = bcp[t * 16 + cl];
#pragma unroll
      for (int v = 0; v < 8; ++v) {
        const int r = rbase + v;
        const _Float16 val = (_Float16)(acc[half][t][v] + bias);
        if (p == 0)      Q16[(size_t)b * N_ * D_ + (size_t)r * D_ + c] = val;
        else if (p == 1) K16[(size_t)b * N_ * D_ + (size_t)r * D_ + c] = val;
        else             Vt16[(size_t)b * D_ * N_ + (size_t)c * N_ + r] = val;
      }
    }
  }
}

// ---------------------------------------------------------------------------
// G2: scores[b,h][n,m] = (Q[b,n,h*64:] . K[b,m,h*64:]) / 8   -> attn buffer
// wave -> 32x64 score tile. grid: y = b*H+h (192), x = 16 blocks (8 waves).
// ---------------------------------------------------------------------------
__global__ void k_scores(const _Float16* __restrict__ Q16,
                         const _Float16* __restrict__ K16,
                         float* __restrict__ attn) {
  const int bh   = blockIdx.y;
  const int b    = bh / H_;
  const int h    = bh % H_;
  const int lane = threadIdx.x & 31;
  const int wid  = blockIdx.x * 8 + (threadIdx.x >> 5);   // 0..127
  const int mt = wid >> 3;   // 0..15 (rows of 32)
  const int nt = wid & 7;    // 0..7  (key-column groups of 64)

  const _Float16* Abase = Q16 + (size_t)b * N_ * D_ + (size_t)mt * 32 * D_ + h * HD_;
  const _Float16* Bbase = K16 + (size_t)b * N_ * D_ + (size_t)nt * 64 * D_ + h * HD_;

  v8f acc[2][4] = {};
#pragma unroll
  for (int k0 = 0; k0 < HD_; k0 += 32) {
    v16h a0 = load_a_f16(Abase + k0, D_, lane);
    v16h a1 = load_a_f16(Abase + (size_t)16 * D_ + k0, D_, lane);
#pragma unroll
    for (int t = 0; t < 4; ++t) {
      v16h bf = load_b_f16(Bbase + (size_t)t * 16 * D_ + k0, D_, lane);
      acc[0][t] = wmma16(a0, bf, acc[0][t]);
      acc[1][t] = wmma16(a1, bf, acc[1][t]);
    }
  }

  float* out = attn + (size_t)bh * N_ * N_;
  const int cl = lane & 15;
#pragma unroll
  for (int half = 0; half < 2; ++half) {
    const int rbase = mt * 32 + half * 16 + (lane >> 4) * 8;
#pragma unroll
    for (int t = 0; t < 4; ++t) {
      const int c = nt * 64 + t * 16 + cl;
#pragma unroll
      for (int v = 0; v < 8; ++v)
        out[(size_t)(rbase + v) * N_ + c] = acc[half][t][v] * 0.125f;
    }
  }
}

// ---------------------------------------------------------------------------
// S: row-wise softmax over attn rows of length 512. one 256-thread block/row.
// ---------------------------------------------------------------------------
__global__ void k_softmax(float* __restrict__ attn) {
  __shared__ float red[256];
  float* p = attn + (size_t)blockIdx.x * N_;
  const int t = threadIdx.x;
  const float a0 = p[t], a1 = p[t + 256];

  red[t] = fmaxf(a0, a1);
  __syncthreads();
  for (int s = 128; s > 0; s >>= 1) {
    if (t < s) red[t] = fmaxf(red[t], red[t + s]);
    __syncthreads();
  }
  const float m = red[0];
  __syncthreads();

  const float e0 = __expf(a0 - m), e1 = __expf(a1 - m);
  red[t] = e0 + e1;
  __syncthreads();
  for (int s = 128; s > 0; s >>= 1) {
    if (t < s) red[t] += red[t + s];
    __syncthreads();
  }
  const float inv = 1.0f / red[0];
  p[t]       = e0 * inv;
  p[t + 256] = e1 * inv;
}

// ---------------------------------------------------------------------------
// G3: ctx[b,n,h*64+d] = sum_m attn[b,h][n,m] * V[b,m,h*64+d]
// A = attn (f32 -> f16 on load), B = Vt (contiguous). wave -> 32x64 tile.
// grid: y = bh (192), x = 2 blocks (8 waves) -> 16 M-tiles of 32 rows.
// ---------------------------------------------------------------------------
__global__ void k_ctx(const float* __restrict__ attn,
                      const _Float16* __restrict__ Vt16,
                      _Float16* __restrict__ ctx16) {
  const int bh   = blockIdx.y;
  const int b    = bh / H_;
  const int h    = bh % H_;
  const int lane = threadIdx.x & 31;
  const int mt   = blockIdx.x * 8 + (threadIdx.x >> 5);   // 0..15

  const float*    Abase = attn + (size_t)bh * N_ * N_ + (size_t)mt * 32 * N_;
  const _Float16* Bbase = Vt16 + (size_t)b * D_ * N_ + (size_t)h * HD_ * N_;

  v8f acc[2][4] = {};
  for (int k0 = 0; k0 < N_; k0 += 32) {
    v16h a0 = load_a_f32cvt(Abase + k0, N_, lane);
    v16h a1 = load_a_f32cvt(Abase + (size_t)16 * N_ + k0, N_, lane);
#pragma unroll
    for (int t = 0; t < 4; ++t) {
      v16h bf = load_b_f16(Bbase + (size_t)t * 16 * N_ + k0, N_, lane);
      acc[0][t] = wmma16(a0, bf, acc[0][t]);
      acc[1][t] = wmma16(a1, bf, acc[1][t]);
    }
  }

  const int cl = lane & 15;
#pragma unroll
  for (int half = 0; half < 2; ++half) {
    const int rbase = mt * 32 + half * 16 + (lane >> 4) * 8;
#pragma unroll
    for (int t = 0; t < 4; ++t) {
      const int c = h * HD_ + t * 16 + cl;
#pragma unroll
      for (int v = 0; v < 8; ++v)
        ctx16[(size_t)b * N_ * D_ + (size_t)(rbase + v) * D_ + c] =
            (_Float16)acc[half][t][v];
    }
  }
}

// ---------------------------------------------------------------------------
// G4: out[b,n,o] = sum_d ctx16[b,n,d] * Wo[o,d] + bo[o]
// wave -> 32x64 tile. grid: 384 blocks of 8 waves.
// ---------------------------------------------------------------------------
__global__ void k_outproj(const _Float16* __restrict__ ctx16,
                          const _Float16* __restrict__ Wo16,
                          const float* __restrict__ bo,
                          float* __restrict__ out) {
  const int lane = threadIdx.x & 31;
  const int wid  = blockIdx.x * 8 + (threadIdx.x >> 5);   // 0..3071
  const int b  = wid / 192;
  const int rm = wid % 192;
  const int mt = rm / 12;              // 0..15 (rows of 32)
  const int nt = rm % 12;              // 0..11 (cols of 64)

  const _Float16* Abase = ctx16 + (size_t)b * N_ * D_ + (size_t)mt * 32 * D_;
  const _Float16* Wbase = Wo16 + (size_t)nt * 64 * D_;

  v8f acc[2][4] = {};
  for (int k0 = 0; k0 < D_; k0 += 32) {
    v16h a0 = load_a_f16(Abase + k0, D_, lane);
    v16h a1 = load_a_f16(Abase + (size_t)16 * D_ + k0, D_, lane);
#pragma unroll
    for (int t = 0; t < 4; ++t) {
      v16h bf = load_b_f16(Wbase + (size_t)t * 16 * D_ + k0, D_, lane);
      acc[0][t] = wmma16(a0, bf, acc[0][t]);
      acc[1][t] = wmma16(a1, bf, acc[1][t]);
    }
  }

  const int cl = lane & 15;
#pragma unroll
  for (int half = 0; half < 2; ++half) {
    const int rbase = mt * 32 + half * 16 + (lane >> 4) * 8;
#pragma unroll
    for (int t = 0; t < 4; ++t) {
      const int c = nt * 64 + t * 16 + cl;
      const float bias = bo[c];
#pragma unroll
      for (int v = 0; v < 8; ++v)
        out[(size_t)b * N_ * D_ + (size_t)(rbase + v) * D_ + c] =
            acc[half][t][v] + bias;
    }
  }
}

// ---------------------------------------------------------------------------
extern "C" void kernel_launch(void* const* d_in, const int* in_sizes, int n_in,
                              void* d_out, int out_size, void* d_ws,
                              size_t ws_size, hipStream_t stream) {
  (void)in_sizes; (void)n_in; (void)out_size; (void)ws_size;

  const float* x  = (const float*)d_in[0];
  const float* g  = (const float*)d_in[1];
  const float* Wq = (const float*)d_in[2];
  const float* bq = (const float*)d_in[3];
  const float* Wk = (const float*)d_in[4];
  const float* bk = (const float*)d_in[5];
  const float* Wv = (const float*)d_in[6];
  const float* bv = (const float*)d_in[7];
  const float* Wo = (const float*)d_in[8];
  const float* bo = (const float*)d_in[9];

  float* outp  = (float*)d_out;                       // [B,N,D]
  float* attnp = outp + (size_t)B_ * N_ * D_;         // [B,H,N,N]

  // workspace carve-up (f16 regions, all well aligned by construction)
  char* w = (char*)d_ws;
  const size_t xe  = (size_t)B_ * N_ * D_;            // 6,291,456
  const size_t wce = (size_t)B_ * D_ * D_;            // 9,437,184
  _Float16* x16  = (_Float16*)(w);                    // xe
  _Float16* Wc16 = (_Float16*)(w + 2 * xe);           // 3*wce
  _Float16* Wo16 = (_Float16*)(w + 2 * (xe + 3 * wce));
  _Float16* Q16  = (_Float16*)(w + 2 * (xe + 3 * wce + (size_t)D_ * D_));
  _Float16* K16  = Q16 + xe;
  _Float16* Vt16 = K16 + xe;                          // stored [b][d][n]
  _Float16* ctx16 = Vt16 + xe;
  float*    bc   = (float*)(ctx16 + xe);              // 3*B*D floats

  // 1) fold gating into per-batch weights / biases, convert x & Wo to f16
  k_combine_w<<<dim3((D_ * D_ / 4 + 255) / 256, 3 * B_), 256, 0, stream>>>(
      g, Wq, Wk, Wv, Wc16);
  k_combine_b<<<(3 * B_ * D_ + 255) / 256, 256, 0, stream>>>(g, bq, bk, bv, bc);
  k_cvt_f16<<<(int)((xe / 4 + 255) / 256), 256, 0, stream>>>(x, x16,
                                                             (int)(xe / 4));
  k_cvt_f16<<<(D_ * D_ / 4 + 255) / 256, 256, 0, stream>>>(Wo, Wo16,
                                                           D_ * D_ / 4);

  // 2) QKV projections (V transposed on store)
  k_qkv_gemm<<<dim3(384, 3), 256, 0, stream>>>(x16, Wc16, bc, Q16, K16, Vt16);

  // 3) scores -> attn region of d_out (scaled)
  k_scores<<<dim3(16, B_ * H_), 256, 0, stream>>>(Q16, K16, attnp);

  // 4) softmax rows in place
  k_softmax<<<B_ * H_ * N_, 256, 0, stream>>>(attnp);

  // 5) ctx = attn @ V
  k_ctx<<<dim3(2, B_ * H_), 256, 0, stream>>>(attnp, Vt16, ctx16);

  // 6) output projection + bias
  k_outproj<<<384, 256, 0, stream>>>(ctx16, Wo16, bo, outp);
}
